// Net_21878563405893
// MI455X (gfx1250) — compile-verified, matrix operand order checked by reading, not run on hardware
//
#include <hip/hip_runtime.h>

// ---------------------------------------------------------------------------
// Shapes (fixed by the reference):
//   B=64, N=128, MAX_LEVEL=3, T(N_TRUNKS)=32, L(TRUNK_LEN)=32, H=256, OUT=10
//   M = B*T = 2048 sequences per level.
// One WG = one (level, batch-row) = 32 sequences (all trunks of that row).
// ---------------------------------------------------------------------------

typedef __attribute__((ext_vector_type(16))) __bf16 v16bf;
typedef __attribute__((ext_vector_type(8)))  float  v8f;

union BFrag { uint4 q[2]; v16bf v; };

__device__ __forceinline__ unsigned short f2bf(float x) {
    unsigned int u = __float_as_uint(x);
    unsigned int r = (u + 0x7FFFu + ((u >> 16) & 1u)) >> 16;   // round-to-nearest-even
    return (unsigned short)r;
}
__device__ __forceinline__ float sigmoidf_(float x) { return 1.0f / (1.0f + __expf(-x)); }
__device__ __forceinline__ float tanhf_(float x)    { return 2.0f / (1.0f + __expf(-2.0f * x)) - 1.0f; }

// ---------------------------------------------------------------------------
// Prep 1: fold the rank-1 encoder into the LSTM input path.
//   u[l][j]  = dot(W_ih[l][j,:], W_enc)
//   vb[l][j] = dot(W_ih[l][j,:], b_enc) + b_ih[l][j] + b_hh[l][j]
// ---------------------------------------------------------------------------
__global__ __launch_bounds__(256) void prep_uv_kernel(
    const float* __restrict__ wenc, const float* __restrict__ benc,
    const float* __restrict__ wih,  const float* __restrict__ bih,
    const float* __restrict__ bhh,  float* __restrict__ u, float* __restrict__ vb)
{
    int idx = blockIdx.x * 256 + threadIdx.x;          // 0 .. 3071  (= 3*1024)
    if (idx >= 3 * 1024) return;
    const float* row = wih + (size_t)idx * 256;
    float su = 0.f, sv = 0.f;
    #pragma unroll 8
    for (int k = 0; k < 256; ++k) {
        float w = row[k];
        su = fmaf(w, wenc[k], su);
        sv = fmaf(w, benc[k], sv);
    }
    u[idx]  = su;
    vb[idx] = sv + bih[idx] + bhh[idx];
}

// ---------------------------------------------------------------------------
// Prep 2: W_hh (3,1024,256) f32 -> bf16 row-major (stays hot in 192MB L2).
// ---------------------------------------------------------------------------
__global__ __launch_bounds__(256) void prep_whh_kernel(
    const float* __restrict__ whh, unsigned short* __restrict__ out, int n)
{
    int i = blockIdx.x * 256 + threadIdx.x;
    if (i < n) out[i] = f2bf(whh[i]);
}

// ---------------------------------------------------------------------------
// Prep 3: score[b][o] = sum_l b_lin[l][o]   (levels then atomically accumulate)
// ---------------------------------------------------------------------------
__global__ __launch_bounds__(256) void init_score_kernel(
    const float* __restrict__ blin, float* __restrict__ score)
{
    int i = blockIdx.x * 256 + threadIdx.x;            // 0 .. 639
    if (i < 64 * 10) {
        int o = i % 10;
        score[i] = blin[o] + blin[10 + o] + blin[20 + o];
    }
}

// ---------------------------------------------------------------------------
// Main: one WG = (level l, batch row b), 32 sequences. 8 waves; wave w owns
// N-tiles w*8..w*8+7 of the 32x1024 gate GEMM as TWO 16-row accumulator sets
// sharing each B fragment (halves L2 traffic vs 16-row tiles — this kernel is
// L2-BW bound: W_hh bf16 streams from L2 every step). 128 WMMA per step/wave.
// Gates drain through a 16-row LDS buffer in two phases to cap LDS at ~125KB
// (2 WGs / WGP co-resident).
// ---------------------------------------------------------------------------
__global__ __launch_bounds__(256) void lstm_level_kernel(
    const float* __restrict__ bfeat,          // (64,128)
    const int*   __restrict__ tidx,           // (3,32,32)
    const int*   __restrict__ tlen,           // (3,32)
    const float* __restrict__ u_all,          // (3,1024)
    const float* __restrict__ vb_all,         // (3,1024)
    const unsigned short* __restrict__ whhbf, // (3,1024,256) bf16
    const float* __restrict__ wlin,           // (3,256,10)
    float*       __restrict__ score)          // (64,10)
{
    const int b    = blockIdx.x;              // 0..63 : batch row == 32 trunks
    const int l    = blockIdx.y;              // 0..2  : level
    const int tid  = threadIdx.x;             // 0..255
    const int lane = tid & 31;
    const int wave = tid >> 5;                // 0..7

    // ---- dynamic LDS carve-up (~125 KB) ----
    extern __shared__ char smem_raw[];
    float* gates = (float*)smem_raw;                     // 16*1024 f32 (64 KB, 2-phase)
    float* c_s   = gates + 16 * 1024;                    // 32*256  f32 (32 KB)
    float* hsum  = c_s + 32 * 256;                       // 256     f32
    float* fseq  = hsum + 256;                           // 32*32   f32 (4 KB)
    float* u_s   = fseq + 1024;                          // 1024    f32
    float* vb_s  = u_s + 1024;                           // 1024    f32
    unsigned short* hbf = (unsigned short*)(vb_s + 1024);// 32*256  bf16 (16 KB)
    int* len_s = (int*)(hbf + 32 * 256);                 // 32

    // ---- init state ----
    for (int i = tid; i < 32 * 256; i += 256) { c_s[i] = 0.f; hbf[i] = 0; }
    hsum[tid] = 0.f;
    for (int i = tid; i < 1024; i += 256) {
        u_s[i]  = u_all[l * 1024 + i];
        vb_s[i] = vb_all[l * 1024 + i];
    }
    // gather scalar input feature per (trunk r, timestep t): rank-1 fold
    for (int e = tid; e < 1024; e += 256) {
        int r = e >> 5, t = e & 31;                      // r == trunk id
        int col = tidx[(l * 32 + r) * 32 + t];
        fseq[r * 32 + t] = bfeat[b * 128 + col];
    }
    if (tid < 32) {
        int lv = tlen[l * 32 + tid];
        len_s[tid] = lv < 1 ? 1 : lv;
    }
    __syncthreads();

    // 16-bit WMMA lane layout: lanes 0-15 hold K in [0,8)+[16,24),
    // lanes 16-31 hold K in [8,16)+[24,32)  -> two contiguous 16B runs.
    const int ma   = lane & 15;               // A: row m / B: col n / D: col n
    const int koff = (lane >> 4) << 3;        // 0 or 8 halves
    const int mhi  = (lane >> 4) << 3;        // D rows: vgpr r -> m = mhi + r
    const unsigned short* whh_l = whhbf + (size_t)l * 1024 * 256;

    for (int t = 0; t < 32; ++t) {
        // Opaque zero in a VGPR: B addresses formally loop-variant so LICM
        // cannot hoist (and spill) the W_hh fragments across timesteps.
        int toff = 0;
        asm volatile("" : "+v"(toff));

        // ==== gates(32x1024) = h(32x256) @ W_hh^T : two 16-row acc sets ====
        v8f acc0[8] = {};
        v8f acc1[8] = {};
        #pragma unroll
        for (int kc = 0; kc < 8; ++kc) {
            BFrag a0, a1;
            const unsigned short* ap0 = &hbf[ma * 256 + kc * 32 + koff];
            const unsigned short* ap1 = &hbf[(16 + ma) * 256 + kc * 32 + koff];
            a0.q[0] = *(const uint4*)(ap0);
            a0.q[1] = *(const uint4*)(ap0 + 16);
            a1.q[0] = *(const uint4*)(ap1);
            a1.q[1] = *(const uint4*)(ap1 + 16);
            #pragma unroll
            for (int i = 0; i < 8; ++i) {
                int j = (wave * 8 + i) * 16 + ma;        // gate column
                const unsigned short* bp =
                    whh_l + ((size_t)j * 256 + (kc * 32 + koff + toff));
                BFrag bb;
                bb.q[0] = *(const uint4*)(bp);
                bb.q[1] = *(const uint4*)(bp + 16);
                acc0[i] = __builtin_amdgcn_wmma_f32_16x16x32_bf16(
                    false, a0.v, false, bb.v, (short)0, acc0[i], false, false);
                acc1[i] = __builtin_amdgcn_wmma_f32_16x16x32_bf16(
                    false, a1.v, false, bb.v, (short)0, acc1[i], false, false);
            }
        }

        // ==== two-phase drain: rows [0,16) then [16,32) =====================
        #pragma unroll
        for (int half = 0; half < 2; ++half) {
            const int rbase = half * 16;
            // scatter D (vgpr r -> row rbase+mhi+r, col tile*16+ma) into LDS
            #pragma unroll
            for (int i = 0; i < 8; ++i) {
                int n = (wave * 8 + i) * 16 + ma;
                #pragma unroll
                for (int r = 0; r < 8; ++r)
                    gates[(mhi + r) * 1024 + n] = half ? acc1[i][r] : acc0[i][r];
            }
            __syncthreads();

            // fused bias + rank-1 input + LSTM cell update for 16 rows
            {
                const int k = tid;                       // hidden unit
                const float us0 = u_s[k],       vs0 = vb_s[k];
                const float us1 = u_s[256 + k], vs1 = vb_s[256 + k];
                const float us2 = u_s[512 + k], vs2 = vb_s[512 + k];
                const float us3 = u_s[768 + k], vs3 = vb_s[768 + k];
                float hcap = 0.f;
                #pragma unroll 4
                for (int r = 0; r < 16; ++r) {
                    const int row = rbase + r;
                    const float ft = fseq[row * 32 + t];
                    const float* g = &gates[r * 1024 + k];
                    float gi = g[0]   + ft * us0 + vs0;  // i
                    float gf = g[256] + ft * us1 + vs1;  // f
                    float gg = g[512] + ft * us2 + vs2;  // g
                    float go = g[768] + ft * us3 + vs3;  // o
                    float c  = sigmoidf_(gf) * c_s[row * 256 + k]
                             + sigmoidf_(gi) * tanhf_(gg);
                    float h  = sigmoidf_(go) * tanhf_(c);
                    c_s[row * 256 + k] = c;
                    hbf[row * 256 + k] = f2bf(h);
                    if (len_s[row] - 1 == t) hcap += h;  // capture h_last
                }
                hsum[k] += hcap;  // only thread k touches hsum[k]
            }
            __syncthreads();
        }
    }

    // ==== score[b][:] += h_trunk @ W_lin[l]  (hsum = sum over all 32 trunks)
    if (tid < 10) {
        const float* wl = wlin + l * 256 * 10 + tid;
        float s = 0.f;
        #pragma unroll 8
        for (int k = 0; k < 256; ++k) s = fmaf(hsum[k], wl[k * 10], s);
        atomicAdd(&score[b * 10 + tid], s);
    }
}

// ---------------------------------------------------------------------------
// Host launcher
// ---------------------------------------------------------------------------
extern "C" void kernel_launch(void* const* d_in, const int* in_sizes, int n_in,
                              void* d_out, int out_size, void* d_ws, size_t ws_size,
                              hipStream_t stream) {
    const float* bfeat = (const float*)d_in[0];   // (64,128)
    const int*   tidx  = (const int*)  d_in[1];   // (3,32,32)
    const int*   tlen  = (const int*)  d_in[2];   // (3,32)
    const float* wenc  = (const float*)d_in[3];   // (1,256)
    const float* benc  = (const float*)d_in[4];   // (256)
    const float* wih   = (const float*)d_in[5];   // (3,1024,256)
    const float* whh   = (const float*)d_in[6];   // (3,1024,256)
    const float* bih   = (const float*)d_in[7];   // (3,1024)
    const float* bhh   = (const float*)d_in[8];   // (3,1024)
    const float* wlin  = (const float*)d_in[9];   // (3,256,10)
    const float* blin  = (const float*)d_in[10];  // (3,10)
    float* score = (float*)d_out;                 // (64,10)

    // workspace: u (12KB) | vb (12KB) | W_hh bf16 (1.5MB)
    char* ws = (char*)d_ws;
    float*          u_dev  = (float*)(ws);
    float*          vb_dev = (float*)(ws + 12288);
    unsigned short* whhbf  = (unsigned short*)(ws + 24576);

    init_score_kernel<<<3, 256, 0, stream>>>(blin, score);
    prep_uv_kernel<<<12, 256, 0, stream>>>(wenc, benc, wih, bih, bhh, u_dev, vb_dev);
    prep_whh_kernel<<<(3 * 1024 * 256 + 255) / 256, 256, 0, stream>>>(whh, whhbf, 3 * 1024 * 256);

    // dynamic LDS: gates(64K) + c(32K) + hsum/fseq/u/vb (~13K) + hbf(16K) + len
    const size_t smem = (16 * 1024 + 32 * 256 + 256 + 32 * 32 + 1024 + 1024) * sizeof(float)
                      + 32 * 256 * sizeof(unsigned short) + 32 * sizeof(int);
    dim3 grid(64, 3);
    lstm_level_kernel<<<grid, 256, smem, stream>>>(bfeat, tidx, tlen, u_dev, vb_dev,
                                                   whhbf, wlin, score);
}
